// MambaCont_49684181680395
// MI455X (gfx1250) — matrix operand order repaired and checked by reading, not compiled, not run on hardware
//
#include <hip/hip_runtime.h>
#include <hip/hip_bf16.h>
#include <stdint.h>

#define Bb    8
#define Ll    1024
#define QUES  4000
#define EMB   256
#define DM    1024
#define DI    2048
#define NS    16
#define DTR   64
#define KC    4
#define MROWS (Bb*Ll)   // 8192

typedef __attribute__((ext_vector_type(16))) __bf16 v16bf;
typedef __attribute__((ext_vector_type(8)))  __bf16 v8bf;
typedef __attribute__((ext_vector_type(8)))  float  v8f;

// ---------------------------------------------------------------------------
// CDNA5 async copy: global -> LDS, 16B per lane, tracked by ASYNCcnt
// ---------------------------------------------------------------------------
__device__ __forceinline__ void async_copy_b128(unsigned lds_addr, unsigned long long gaddr) {
  asm volatile("global_load_async_to_lds_b128 %0, %1, off"
               :: "v"(lds_addr), "v"(gaddr) : "memory");
}
__device__ __forceinline__ unsigned lds_off(const void* p) {
  // flat LDS-aperture address: low 32 bits == LDS byte offset
  return (unsigned)(uintptr_t)p;
}

// ---------------------------------------------------------------------------
// Generic fp32 -> bf16 conversion
// ---------------------------------------------------------------------------
__global__ __launch_bounds__(256) void f32_to_bf16_kernel(const float* __restrict__ in,
                                                          __bf16* __restrict__ out,
                                                          long long n) {
  long long i = (long long)blockIdx.x * 256 + threadIdx.x;
  if (i < n) out[i] = (__bf16)in[i];
}

// ---------------------------------------------------------------------------
// Embedding gather + concat + LayerNorm0 (one block per token row)
// ---------------------------------------------------------------------------
__global__ __launch_bounds__(256) void embed_ln0_kernel(
    const int* __restrict__ x, const int* __restrict__ qid,
    const float* __restrict__ emb, const float* __restrict__ qc,
    const float* __restrict__ g, const float* __restrict__ b,
    float* __restrict__ itemf, __bf16* __restrict__ itembf) {
  const int row = blockIdx.x;
  const int tid = threadIdx.x;
  const int xv = x[row];
  const int qv = qid[row];
  float vals[4];
#pragma unroll
  for (int j = 0; j < 4; ++j) {
    int c = tid + j * 256;
    vals[j] = (c < EMB) ? emb[(long long)xv * EMB + c]
                        : qc[(long long)qv * 768 + (c - EMB)];
  }
  __shared__ float red[256];
  float s = vals[0] + vals[1] + vals[2] + vals[3];
  red[tid] = s; __syncthreads();
  for (int st = 128; st > 0; st >>= 1) { if (tid < st) red[tid] += red[tid + st]; __syncthreads(); }
  const float mean = red[0] * (1.0f / DM);
  __syncthreads();
  float ls = 0.f;
#pragma unroll
  for (int j = 0; j < 4; ++j) { float d = vals[j] - mean; ls += d * d; }
  red[tid] = ls; __syncthreads();
  for (int st = 128; st > 0; st >>= 1) { if (tid < st) red[tid] += red[tid + st]; __syncthreads(); }
  const float inv = rsqrtf(red[0] * (1.0f / DM) + 1e-12f);
#pragma unroll
  for (int j = 0; j < 4; ++j) {
    int c = tid + j * 256;
    float o = (vals[j] - mean) * inv * g[c] + b[c];
    itemf[(long long)row * DM + c]  = o;
    itembf[(long long)row * DM + c] = (__bf16)o;
  }
}

// ---------------------------------------------------------------------------
// WMMA GEMM:  C[M,N] = A[M,K] (bf16 row-major) x W[N,K]^T (bf16 row-major)
// 128x128 tile / 256-thread block; 8 waves as 4(M) x 2(N); wave = 32x64 slab
// = 8 x v_wmma_f32_16x16x32_bf16 per K-step. Async global->LDS double buffer.
// EPI: 0=none  1=softplus(v+bias)  2=sigmoid(v+bias)
// ---------------------------------------------------------------------------
template<int EPI>
__device__ __forceinline__ float epi_apply(float v, const float* __restrict__ bias, int col) {
  if (EPI == 1) { float xx = v + bias[col]; return xx > 20.f ? xx : log1pf(__expf(xx)); }
  if (EPI == 2) { float xx = v + bias[col]; return 1.f / (1.f + __expf(-xx)); }
  return v;
}

template<int EPI>
__global__ __launch_bounds__(256) void gemm_wmma_kernel(
    const __bf16* __restrict__ A, const __bf16* __restrict__ W,
    const float* __restrict__ bias, float* __restrict__ C,
    int M, int N, int K, int ldc) {
  __shared__ __bf16 As[2][128][40];   // +8 bf16 pad per row for bank spread
  __shared__ __bf16 Bs[2][128][40];

  const int tid  = threadIdx.x;
  const int lane = tid & 31;
  const int wave = tid >> 5;
  const int wm   = wave >> 1;          // 0..3  -> M base wm*32
  const int wn   = wave & 1;           // 0..1  -> N base wn*64
  const int bm   = blockIdx.y * 128;
  const int bn   = blockIdx.x * 128;
  const int half = lane >> 4;          // 0/1
  const int l15  = lane & 15;

  // staging assignment: 512 chunks of 16B per 128x32 tile; 2 chunks/thread
  const int crow  = tid >> 1;                  // 0..127
  const int ccol0 = (tid & 1) ? 8 : 0;         // bf16 units
  const int ccol1 = ccol0 + 16;

  // clamp OOB B rows to a valid row: garbage only lands in store-guarded cols
  int gnB = bn + crow; if (gnB >= N) gnB = N - 1;
  const __bf16* Arow = A + (long long)(bm + crow) * K;
  const __bf16* Brow = W + (long long)gnB * K;

  v8f acc[2][4];
#pragma unroll
  for (int mt = 0; mt < 2; ++mt)
#pragma unroll
    for (int nt = 0; nt < 4; ++nt) acc[mt][nt] = (v8f){};

  const int nstages = K >> 5;

  auto stage = [&](int buf, int k0) {
    async_copy_b128(lds_off(&As[buf][crow][ccol0]),
                    (unsigned long long)(uintptr_t)(Arow + k0 + ccol0));
    async_copy_b128(lds_off(&As[buf][crow][ccol1]),
                    (unsigned long long)(uintptr_t)(Arow + k0 + ccol1));
    async_copy_b128(lds_off(&Bs[buf][crow][ccol0]),
                    (unsigned long long)(uintptr_t)(Brow + k0 + ccol0));
    async_copy_b128(lds_off(&Bs[buf][crow][ccol1]),
                    (unsigned long long)(uintptr_t)(Brow + k0 + ccol1));
  };

  stage(0, 0);

  for (int s = 0; s < nstages; ++s) {
    const int buf = s & 1;
    const int k0n = (s + 1) << 5;
    if (s + 1 < nstages) {
      stage(buf ^ 1, k0n);
      // warm L2 two stages out
      if (s + 2 < nstages) __builtin_prefetch(Arow + ((s + 2) << 5), 0, 0);
      asm volatile("s_wait_asynccnt 0x4" ::: "memory");  // current stage retired
    } else {
      asm volatile("s_wait_asynccnt 0x0" ::: "memory");
    }
    __syncthreads();

    // ---- fragments (documented CDNA5 bf16 VGPR layouts) ----
    v16bf af[2], bfr[4];
#pragma unroll
    for (int mt = 0; mt < 2; ++mt) {
      const int m = wm * 32 + mt * 16 + l15;
      v8bf a0 = *(const v8bf*)&As[buf][m][half * 8];
      v8bf a1 = *(const v8bf*)&As[buf][m][16 + half * 8];
      af[mt] = __builtin_shufflevector(a0, a1, 0,1,2,3,4,5,6,7,8,9,10,11,12,13,14,15);
    }
#pragma unroll
    for (int nt = 0; nt < 4; ++nt) {
      const int n = wn * 64 + nt * 16 + l15;
      v8bf b0 = *(const v8bf*)&Bs[buf][n][half * 8];
      v8bf b1 = *(const v8bf*)&Bs[buf][n][16 + half * 8];
      bfr[nt] = __builtin_shufflevector(b0, b1, 0,1,2,3,4,5,6,7,8,9,10,11,12,13,14,15);
    }
#pragma unroll
    for (int mt = 0; mt < 2; ++mt)
#pragma unroll
      for (int nt = 0; nt < 4; ++nt)
        acc[mt][nt] = __builtin_amdgcn_wmma_f32_16x16x32_bf16(
            false, af[mt], false, bfr[nt], (short)0, acc[mt][nt], false, false);
    __syncthreads();
  }

  // ---- store: C/D layout -> VGPR r: M = r + half*8, N = l15 ----
#pragma unroll
  for (int mt = 0; mt < 2; ++mt) {
    const int rowb = bm + wm * 32 + mt * 16 + half * 8;
#pragma unroll
    for (int nt = 0; nt < 4; ++nt) {
      const int col = bn + wn * 64 + nt * 16 + l15;
      if (col < N) {
#pragma unroll
        for (int r = 0; r < 8; ++r)
          C[(long long)(rowb + r) * ldc + col] = epi_apply<EPI>(acc[mt][nt][r], bias, col);
      }
    }
  }
}

// ---------------------------------------------------------------------------
// Depthwise causal conv (K=4) + bias + SiLU.  u = cols[0..DI) of u_r (ld 4096)
// ---------------------------------------------------------------------------
__global__ __launch_bounds__(256) void conv_silu_kernel(
    const float* __restrict__ u_r, const float* __restrict__ w,
    const float* __restrict__ cb, float* __restrict__ uact,
    __bf16* __restrict__ uactbf) {
  long long idx = (long long)blockIdx.x * 256 + threadIdx.x;
  if (idx >= (long long)MROWS * DI) return;
  const int d   = (int)(idx % DI);
  const long long row = idx / DI;
  const int t   = (int)(row % Ll);
  float acc = cb[d];
#pragma unroll
  for (int k = 0; k < KC; ++k) {
    int back = KC - 1 - k;
    if (t - back >= 0)
      acc += u_r[(row - back) * (2LL * DI) + d] * w[d * KC + k];
  }
  float s = acc / (1.f + __expf(-acc));
  uact[idx]   = s;
  uactbf[idx] = (__bf16)s;
}

// ---------------------------------------------------------------------------
// dt slice (first 64 of 96 cols of dbl) -> bf16
// ---------------------------------------------------------------------------
__global__ __launch_bounds__(64) void extract_dt_kernel(const float* __restrict__ dbl,
                                                        __bf16* __restrict__ dtbf) {
  int i = blockIdx.x, c = threadIdx.x;
  dtbf[(long long)i * DTR + c] = (__bf16)dbl[(long long)i * (DTR + 2 * NS) + c];
}

// ---------------------------------------------------------------------------
// Selective scan: one thread per (b,d); h[16] in registers; B/C via LDS
// ---------------------------------------------------------------------------
__global__ __launch_bounds__(256) void scan_kernel(
    const float* __restrict__ delta, const float* __restrict__ dbl,
    const float* __restrict__ uact, const float* __restrict__ A_log,
    float* __restrict__ y) {
  const int b = blockIdx.x >> 3;
  const int d = (blockIdx.x & 7) * 256 + threadIdx.x;
  float Av[NS], h[NS];
#pragma unroll
  for (int n = 0; n < NS; ++n) { Av[n] = -__expf(A_log[(long long)d * NS + n]); h[n] = 0.f; }
  __shared__ float Bs[NS], Cs[NS];
  for (int t = 0; t < Ll; ++t) {
    const long long row = (long long)b * Ll + t;
    if (threadIdx.x < 32) {
      float v = dbl[row * (DTR + 2 * NS) + DTR + threadIdx.x];
      if (threadIdx.x < NS) Bs[threadIdx.x] = v; else Cs[threadIdx.x - NS] = v;
    }
    __syncthreads();
    const float dlt = delta[row * DI + d];
    const float ut  = uact[row * DI + d];
    float acc = 0.f;
#pragma unroll
    for (int n = 0; n < NS; ++n) {
      float dA = __expf(dlt * Av[n]);
      h[n] = dA * h[n] + dlt * Bs[n] * ut;
      acc += h[n] * Cs[n];
    }
    y[row * DI + d] = acc;
    __syncthreads();
  }
}

// ---------------------------------------------------------------------------
// Gating: (y + u*D) * silu(r) -> bf16     (r = cols [DI..2*DI) of u_r)
// ---------------------------------------------------------------------------
__global__ __launch_bounds__(256) void gate_kernel(
    const float* __restrict__ y, const float* __restrict__ uact,
    const float* __restrict__ u_r, const float* __restrict__ D,
    __bf16* __restrict__ out) {
  long long idx = (long long)blockIdx.x * 256 + threadIdx.x;
  if (idx >= (long long)MROWS * DI) return;
  const int d = (int)(idx % DI);
  const long long row = idx / DI;
  const float r  = u_r[row * (2LL * DI) + DI + d];
  const float sr = r / (1.f + __expf(-r));
  out[idx] = (__bf16)((y[idx] + uact[idx] * D[d]) * sr);
}

// ---------------------------------------------------------------------------
// LayerNorm1 over (h + item) -> bf16
// ---------------------------------------------------------------------------
__global__ __launch_bounds__(256) void ln1_kernel(
    const float* __restrict__ hbuf, const float* __restrict__ itemf,
    const float* __restrict__ g, const float* __restrict__ b,
    __bf16* __restrict__ out) {
  const int row = blockIdx.x;
  const int tid = threadIdx.x;
  float vals[4];
#pragma unroll
  for (int j = 0; j < 4; ++j) {
    int c = tid + j * 256;
    vals[j] = hbuf[(long long)row * DM + c] + itemf[(long long)row * DM + c];
  }
  __shared__ float red[256];
  float s = vals[0] + vals[1] + vals[2] + vals[3];
  red[tid] = s; __syncthreads();
  for (int st = 128; st > 0; st >>= 1) { if (tid < st) red[tid] += red[tid + st]; __syncthreads(); }
  const float mean = red[0] * (1.0f / DM);
  __syncthreads();
  float ls = 0.f;
#pragma unroll
  for (int j = 0; j < 4; ++j) { float dd = vals[j] - mean; ls += dd * dd; }
  red[tid] = ls; __syncthreads();
  for (int st = 128; st > 0; st >>= 1) { if (tid < st) red[tid] += red[tid + st]; __syncthreads(); }
  const float inv = rsqrtf(red[0] * (1.0f / DM) + 1e-12f);
#pragma unroll
  for (int j = 0; j < 4; ++j) {
    int c = tid + j * 256;
    out[(long long)row * DM + c] = (__bf16)((vals[j] - mean) * inv * g[c] + b[c]);
  }
}

// ---------------------------------------------------------------------------
// Host launcher
// ---------------------------------------------------------------------------
static inline size_t align256(size_t x) { return (x + 255) & ~(size_t)255; }

extern "C" void kernel_launch(void* const* d_in, const int* in_sizes, int n_in,
                              void* d_out, int out_size, void* d_ws, size_t ws_size,
                              hipStream_t stream) {
  const int*   x        = (const int*)d_in[0];
  const int*   q_id     = (const int*)d_in[1];
  const float* emb      = (const float*)d_in[2];
  const float* qc       = (const float*)d_in[3];
  const float* ln0_g    = (const float*)d_in[4];
  const float* ln0_b    = (const float*)d_in[5];
  const float* in_proj  = (const float*)d_in[6];
  const float* conv_w   = (const float*)d_in[7];
  const float* conv_b   = (const float*)d_in[8];
  const float* x_proj   = (const float*)d_in[9];
  const float* dt_proj  = (const float*)d_in[10];
  const float* dt_pb    = (const float*)d_in[11];
  const float* A_log    = (const float*)d_in[12];
  const float* D_param  = (const float*)d_in[13];
  const float* out_proj = (const float*)d_in[14];
  const float* ln1_g    = (const float*)d_in[15];
  const float* ln1_b    = (const float*)d_in[16];
  const float* fc_w     = (const float*)d_in[17];
  const float* fc_b     = (const float*)d_in[18];
  float* out = (float*)d_out;

  char* ws = (char*)d_ws;
  size_t off = 0;
  auto alloc = [&](size_t bytes) -> char* { char* p = ws + off; off = align256(off + bytes); return p; };

  float*  itemf   = (float*) alloc((size_t)MROWS * DM * 4);
  __bf16* itembf  = (__bf16*)alloc((size_t)MROWS * DM * 2);
  __bf16* w_in    = (__bf16*)alloc((size_t)2 * DI * DM * 2);
  float*  u_r     = (float*) alloc((size_t)MROWS * 2 * DI * 4);
  float*  uact    = (float*) alloc((size_t)MROWS * DI * 4);
  __bf16* uactbf  = (__bf16*)alloc((size_t)MROWS * DI * 2);
  __bf16* w_xp    = (__bf16*)alloc((size_t)(DTR + 2 * NS) * DI * 2);
  float*  dbl     = (float*) alloc((size_t)MROWS * (DTR + 2 * NS) * 4);
  __bf16* dtbf    = (__bf16*)alloc((size_t)MROWS * DTR * 2);
  __bf16* w_dt    = (__bf16*)alloc((size_t)DI * DTR * 2);
  float*  delta   = (float*) alloc((size_t)MROWS * DI * 4);
  float*  yb      = (float*) alloc((size_t)MROWS * DI * 4);
  __bf16* ygbf    = (__bf16*)alloc((size_t)MROWS * DI * 2);
  __bf16* w_out   = (__bf16*)alloc((size_t)DM * DI * 2);
  float*  hbuf    = (float*) alloc((size_t)MROWS * DM * 4);
  __bf16* lnbf    = (__bf16*)alloc((size_t)MROWS * DM * 2);
  __bf16* w_fc    = (__bf16*)alloc((size_t)QUES * DM * 2);
  (void)ws_size; (void)n_in; (void)in_sizes; (void)out_size;

  auto cvt = [&](const float* src, __bf16* dst, long long n) {
    f32_to_bf16_kernel<<<(unsigned)((n + 255) / 256), 256, 0, stream>>>(src, dst, n);
  };

  // 1. embed + concat + LN0
  embed_ln0_kernel<<<MROWS, 256, 0, stream>>>(x, q_id, emb, qc, ln0_g, ln0_b, itemf, itembf);

  // 2. in_proj GEMM: (8192 x 4096 x 1024)
  cvt(in_proj, w_in, (long long)2 * DI * DM);
  gemm_wmma_kernel<0><<<dim3((2 * DI) / 128, MROWS / 128), 256, 0, stream>>>(
      itembf, w_in, nullptr, u_r, MROWS, 2 * DI, DM, 2 * DI);

  // 3. depthwise conv + SiLU
  conv_silu_kernel<<<(unsigned)(((long long)MROWS * DI + 255) / 256), 256, 0, stream>>>(
      u_r, conv_w, conv_b, uact, uactbf);

  // 4. x_proj GEMM: (8192 x 96 x 2048)
  cvt(x_proj, w_xp, (long long)(DTR + 2 * NS) * DI);
  gemm_wmma_kernel<0><<<dim3(1, MROWS / 128), 256, 0, stream>>>(
      uactbf, w_xp, nullptr, dbl, MROWS, DTR + 2 * NS, DI, DTR + 2 * NS);

  // 5. dt_proj GEMM + softplus: (8192 x 2048 x 64)
  extract_dt_kernel<<<MROWS, 64, 0, stream>>>(dbl, dtbf);
  cvt(dt_proj, w_dt, (long long)DI * DTR);
  gemm_wmma_kernel<1><<<dim3(DI / 128, MROWS / 128), 256, 0, stream>>>(
      dtbf, w_dt, dt_pb, delta, MROWS, DI, DTR, DI);

  // 6. selective scan
  scan_kernel<<<Bb * (DI / 256), 256, 0, stream>>>(delta, dbl, uact, A_log, yb);

  // 7. gating
  gate_kernel<<<(unsigned)(((long long)MROWS * DI + 255) / 256), 256, 0, stream>>>(
      yb, uact, u_r, D_param, ygbf);

  // 8. out_proj GEMM: (8192 x 1024 x 2048)
  cvt(out_proj, w_out, (long long)DM * DI);
  gemm_wmma_kernel<0><<<dim3(DM / 128, MROWS / 128), 256, 0, stream>>>(
      ygbf, w_out, nullptr, hbuf, MROWS, DM, DI, DM);

  // 9. LN1 over residual
  ln1_kernel<<<MROWS, 256, 0, stream>>>(hbuf, itemf, ln1_g, ln1_b, lnbf);

  // 10. fc GEMM + bias + sigmoid -> d_out: (8192 x 4000 x 1024)
  cvt(fc_w, w_fc, (long long)QUES * DM);
  gemm_wmma_kernel<2><<<dim3((QUES + 127) / 128, MROWS / 128), 256, 0, stream>>>(
      lnbf, w_fc, fc_b, out, MROWS, QUES, DM, QUES);
}
